// NodeModel_38113539784806
// MI455X (gfx1250) — matrix operand order repaired and checked by reading, not compiled, not run on hardware
//
#include <hip/hip_runtime.h>

typedef __attribute__((ext_vector_type(16))) _Float16 v16h;
typedef __attribute__((ext_vector_type(8)))  _Float16 v8h;
typedef __attribute__((ext_vector_type(8)))  float    v8f;

typedef __attribute__((address_space(3))) char lds_char_t;

// LDS byte offset (within the workgroup's LDS allocation) of a generic pointer.
__device__ __forceinline__ unsigned lds_off_u32(const void* p) {
  return (unsigned)(unsigned long long)(const lds_char_t*)p;
}

// Async DMA: 16 bytes global -> LDS, tracked by ASYNCcnt (CDNA5).
__device__ __forceinline__ void async_load_b128(unsigned lds_off, const void* gaddr) {
  asm volatile("global_load_async_to_lds_b128 %0, %1, off"
               :: "v"(lds_off), "v"(gaddr)
               : "memory");
}

__device__ __forceinline__ void wait_async_zero() {
  asm volatile("s_wait_asynccnt 0x0" ::: "memory");
}

// ---------------------------------------------------------------------------
// WMMA fragment loaders (CDNA5 16x16x32 f16 layouts, wave32)
// A (16x32): lane L holds row m=L&15; elems 0..7 -> K=kb..kb+7, elems 8..15 ->
//            K=16+kb..16+kb+7 where kb=(L>>4)*8.
// B (32x16): lane L holds col n=L&15; same K order as A.
// C/D f32  : vgpr r, lane L -> (m = r + 8*(L>>4), n = L&15).
// ---------------------------------------------------------------------------
__device__ __forceinline__ v16h load_a_lds(const _Float16* sA, int ld,
                                           int rowbase, int kt, int lane) {
  const int m  = lane & 15;
  const int kb = (lane >> 4) * 8;
  const _Float16* p = sA + (rowbase + m) * ld + kt * 32 + kb;
  v8h lo = *(const v8h*)p;         // K = kt*32 + kb .. +7
  v8h hi = *(const v8h*)(p + 16);  // K = kt*32 + 16 + kb .. +7
  v16h r;
#pragma unroll
  for (int i = 0; i < 8; ++i) { r[i] = lo[i]; r[i + 8] = hi[i]; }
  return r;
}

// Weights are pre-swizzled so each lane's 16 halves are contiguous (32B).
__device__ __forceinline__ v16h load_b_sw(const _Float16* W, int KT, int j,
                                          int kt, int lane) {
  return *(const v16h*)(W + ((((j * KT + kt) * 32) + lane) << 4));
}

// ---------------------------------------------------------------------------
// Elementwise f32 -> f16 conversion (x, edge_attr, u pre-pass).
// ---------------------------------------------------------------------------
__global__ void f32_to_f16_kernel(const float* __restrict__ src,
                                  _Float16* __restrict__ dst, long long n) {
  long long i = (long long)blockIdx.x * blockDim.x + threadIdx.x;
  if (i < n) dst[i] = (_Float16)src[i];
}

// ---------------------------------------------------------------------------
// f32 -> f16 weight conversion with B-fragment swizzle. W is [K, Nc] row-major.
// ---------------------------------------------------------------------------
__global__ void convert_swizzle_kernel(const float* __restrict__ W,
                                       _Float16* __restrict__ Wsw,
                                       int K, int Nc) {
  int idx = blockIdx.x * blockDim.x + threadIdx.x;
  if (idx >= K * Nc) return;
  int k = idx / Nc, n = idx % Nc;
  int KT = K >> 5;
  int j  = n >> 4;
  int kt = k >> 5;
  int kr = k & 31;
  int L  = (((kr >> 3) & 1) << 4) | (n & 15);
  int i  = (kr & 7) + ((kr >> 4) << 3);
  Wsw[((((j * KT + kt) * 32) + L) << 4) + i] = (_Float16)W[idx];
}

// ---------------------------------------------------------------------------
// Per-edge in-degree counts (for scatter-mean denominator).
// ---------------------------------------------------------------------------
__global__ void counts_kernel(const int* __restrict__ eidx,
                              float* __restrict__ counts, int E) {
  int e = blockIdx.x * blockDim.x + threadIdx.x;
  if (e < E) atomicAdd(counts + eidx[E + e], 1.0f);
}

// ---------------------------------------------------------------------------
// Edge message MLP + weighted scatter-add.
// 64 edges / block, 4 waves, each wave owns a 16-row strip.
// Inputs staged via async global->LDS DMA (f16 pre-converted).
// ---------------------------------------------------------------------------
__global__ __launch_bounds__(128) void edge_msg_kernel(
    const _Float16* __restrict__ xh, const int* __restrict__ eidx,
    const _Float16* __restrict__ eah, const float* __restrict__ wts,
    const _Float16* __restrict__ W1, const float* __restrict__ b1,
    const float* __restrict__ g1, const float* __restrict__ be1,
    const _Float16* __restrict__ W2, const float* __restrict__ b2,
    float* __restrict__ sums, int E) {
  extern __shared__ char smem[];
  _Float16* sA  = (_Float16*)smem;           // [64][256] gathered input (f16)
  _Float16* sH  = sA + 64 * 256;             // [64][256] hidden (f16)
  float*    sMu = (float*)(sH + 64 * 256);   // [64]
  float*    sRs = sMu + 64;                  // [64]
  int*      sRow = (int*)(sRs + 64);         // [64]
  int*      sCol = sRow + 64;                // [64]
  float*    sWt  = (float*)(sCol + 64);      // [64]

  const int tid  = threadIdx.x;
  const int lane = tid & 31;
  const int rb   = (tid >> 5) * 16;          // wave's row base
  const long long e0 = (long long)blockIdx.x * 64;

  for (int r = tid; r < 64; r += 128) {
    long long e = e0 + r;
    bool valid = e < (long long)E;
    sRow[r] = valid ? eidx[e] : 0;
    sCol[r] = valid ? eidx[(long long)E + e] : -1;
    sWt[r]  = valid ? wts[e] : 0.f;
  }
  __syncthreads();

  // Async-DMA stage A = [xh[row] || eah[e]]: 64 rows x 512B = 2048 x 16B.
  {
    const unsigned sA_base = lds_off_u32(sA);
    for (int ci = tid; ci < 2048; ci += 128) {
      int r  = ci >> 5;
      int ob = (ci & 31) << 4;           // byte offset within 512B f16 row
      long long e = e0 + r;
      long long ec = (e < (long long)E) ? e : (long long)(E - 1);  // clamp tail
      const char* src = (ob < 256)
          ? (const char*)(xh + (long long)sRow[r] * 128) + ob
          : (const char*)(eah + ec * 128) + (ob - 256);
      async_load_b128(sA_base + (unsigned)(r * 512 + ob), src);
    }
    wait_async_zero();
  }
  __syncthreads();

  // Layer 1: [64,256] @ [256,256] + b1, ReLU -> sH
  for (int j = 0; j < 16; ++j) {
    float bv = b1[j * 16 + (lane & 15)];
    v8f c = {bv, bv, bv, bv, bv, bv, bv, bv};
    __builtin_prefetch((const void*)(W1 + (((j * 8) * 32 + lane) << 4)), 0, 0);
#pragma unroll
    for (int kt = 0; kt < 8; ++kt) {
      v16h a = load_a_lds(sA, 256, rb, kt, lane);
      v16h b = load_b_sw(W1, 8, j, kt, lane);
      c = __builtin_amdgcn_wmma_f32_16x16x32_f16(false, a, false, b,
                                                 (short)0, c, false, false);
    }
    int n  = j * 16 + (lane & 15);
    int mo = (lane >> 4) * 8;
#pragma unroll
    for (int r = 0; r < 8; ++r) {
      float v = c[r] > 0.f ? c[r] : 0.f;              // ReLU
      sH[(rb + mo + r) * 256 + n] = (_Float16)v;
    }
  }
  __syncthreads();

  // LayerNorm over 256 features per row.
  for (int r = tid; r < 64; r += 128) {
    const _Float16* hr = sH + r * 256;
    float s = 0.f, s2 = 0.f;
    for (int c0 = 0; c0 < 256; ++c0) {
      float v = (float)hr[c0]; s += v; s2 += v * v;
    }
    float mu  = s * (1.f / 256.f);
    float var = s2 * (1.f / 256.f) - mu * mu;
    sMu[r] = mu;
    sRs[r] = rsqrtf(var + 1e-5f);
  }
  __syncthreads();
  for (int s = tid; s < 64 * 256; s += 128) {
    int r = s >> 8, c0 = s & 255;
    float v = ((float)sH[s] - sMu[r]) * sRs[r] * g1[c0] + be1[c0];
    sH[s] = (_Float16)v;
  }
  __syncthreads();

  // Layer 2: [64,256] @ [256,256] + b2, then weighted scatter-add to sums.
  for (int j = 0; j < 16; ++j) {
    float bv = b2[j * 16 + (lane & 15)];
    v8f c = {bv, bv, bv, bv, bv, bv, bv, bv};
#pragma unroll
    for (int kt = 0; kt < 8; ++kt) {
      v16h a = load_a_lds(sH, 256, rb, kt, lane);
      v16h b = load_b_sw(W2, 8, j, kt, lane);
      c = __builtin_amdgcn_wmma_f32_16x16x32_f16(false, a, false, b,
                                                 (short)0, c, false, false);
    }
    int n  = j * 16 + (lane & 15);
    int mo = (lane >> 4) * 8;
#pragma unroll
    for (int r = 0; r < 8; ++r) {
      int rr  = rb + mo + r;
      int col = sCol[rr];
      if (col >= 0)
        atomicAdd(sums + (long long)col * 256 + n, c[r] * sWt[rr]);
    }
  }
}

// ---------------------------------------------------------------------------
// Node MLP: [x || received || u[batch]] (448) -> 512 (ReLU,LN) -> 384.
// 64 nodes / block, 4 waves. x/u segments DMA'd async; received computed.
// ---------------------------------------------------------------------------
__global__ __launch_bounds__(128) void node_mlp_kernel(
    const _Float16* __restrict__ xh, const float* __restrict__ sums,
    const float* __restrict__ counts, const _Float16* __restrict__ uh,
    const int* __restrict__ node_batch,
    const _Float16* __restrict__ W1, const float* __restrict__ b1,
    const float* __restrict__ g1, const float* __restrict__ be1,
    const _Float16* __restrict__ W2, const float* __restrict__ b2,
    float* __restrict__ out, int N) {
  extern __shared__ char smem[];
  _Float16* sIn = (_Float16*)smem;            // [64][448]
  _Float16* sH  = sIn + 64 * 448;             // [64][512]
  float*    sMu = (float*)(sH + 64 * 512);    // [64]
  float*    sRs = sMu + 64;                   // [64]

  const int tid  = threadIdx.x;
  const int lane = tid & 31;
  const int rb   = (tid >> 5) * 16;
  const long long n0 = (long long)blockIdx.x * 64;

  const unsigned sIn_base = lds_off_u32(sIn);

  // x segment: cols 0..127 (256B/row) via async DMA. 64 rows x 16 chunks.
  for (int ci = tid; ci < 1024; ci += 128) {
    int r  = ci >> 4;
    int ob = (ci & 15) << 4;
    long long node = n0 + r;
    if (node >= (long long)N) node = N - 1;   // clamp tail (masked at store)
    async_load_b128(sIn_base + (unsigned)(r * 896 + ob),
                    (const char*)(xh + node * 128) + ob);
  }
  // u segment: cols 384..447 (128B/row at byte offset 768). 64 rows x 8 chunks.
  for (int ci = tid; ci < 512; ci += 128) {
    int r  = ci >> 3;
    int ob = (ci & 7) << 4;
    long long node = n0 + r;
    if (node >= (long long)N) node = N - 1;
    int b = node_batch[node];
    async_load_b128(sIn_base + (unsigned)(r * 896 + 768 + ob),
                    (const char*)(uh + (long long)b * 64) + ob);
  }
  // received segment: cols 128..383 = sums/max(counts,1) (needs arithmetic).
  for (int s = tid; s < 64 * 64; s += 128) {
    int r  = s >> 6;
    int c4 = (s & 63) << 2;
    long long node = n0 + r;
    float v0 = 0.f, v1 = 0.f, v2 = 0.f, v3 = 0.f;
    if (node < (long long)N) {
      float inv = 1.f / fmaxf(counts[node], 1.f);
      float4 v = *(const float4*)(sums + node * 256 + c4);
      v0 = v.x * inv; v1 = v.y * inv; v2 = v.z * inv; v3 = v.w * inv;
    }
    _Float16* d = sIn + r * 448 + 128 + c4;
    d[0] = (_Float16)v0; d[1] = (_Float16)v1;
    d[2] = (_Float16)v2; d[3] = (_Float16)v3;
  }
  wait_async_zero();
  __syncthreads();

  // Layer 1: [64,448] @ [448,512] + b1, ReLU -> sH
  for (int j = 0; j < 32; ++j) {
    float bv = b1[j * 16 + (lane & 15)];
    v8f c = {bv, bv, bv, bv, bv, bv, bv, bv};
    __builtin_prefetch((const void*)(W1 + (((j * 14) * 32 + lane) << 4)), 0, 0);
#pragma unroll
    for (int kt = 0; kt < 14; ++kt) {
      v16h a = load_a_lds(sIn, 448, rb, kt, lane);
      v16h b = load_b_sw(W1, 14, j, kt, lane);
      c = __builtin_amdgcn_wmma_f32_16x16x32_f16(false, a, false, b,
                                                 (short)0, c, false, false);
    }
    int n  = j * 16 + (lane & 15);
    int mo = (lane >> 4) * 8;
#pragma unroll
    for (int r = 0; r < 8; ++r) {
      float v = c[r] > 0.f ? c[r] : 0.f;  // ReLU
      sH[(rb + mo + r) * 512 + n] = (_Float16)v;
    }
  }
  __syncthreads();

  // LayerNorm over 512 features per row.
  for (int r = tid; r < 64; r += 128) {
    const _Float16* hr = sH + r * 512;
    float s = 0.f, s2 = 0.f;
    for (int c0 = 0; c0 < 512; ++c0) {
      float v = (float)hr[c0]; s += v; s2 += v * v;
    }
    float mu  = s * (1.f / 512.f);
    float var = s2 * (1.f / 512.f) - mu * mu;
    sMu[r] = mu;
    sRs[r] = rsqrtf(var + 1e-5f);
  }
  __syncthreads();
  for (int s = tid; s < 64 * 512; s += 128) {
    int r = s >> 9, c0 = s & 511;
    float v = ((float)sH[s] - sMu[r]) * sRs[r] * g1[c0] + be1[c0];
    sH[s] = (_Float16)v;
  }
  __syncthreads();

  // Layer 2: [64,512] @ [512,384] + b2 -> out (f32)
  for (int j = 0; j < 24; ++j) {
    float bv = b2[j * 16 + (lane & 15)];
    v8f c = {bv, bv, bv, bv, bv, bv, bv, bv};
#pragma unroll
    for (int kt = 0; kt < 16; ++kt) {
      v16h a = load_a_lds(sH, 512, rb, kt, lane);
      v16h b = load_b_sw(W2, 16, j, kt, lane);
      c = __builtin_amdgcn_wmma_f32_16x16x32_f16(false, a, false, b,
                                                 (short)0, c, false, false);
    }
    int n  = j * 16 + (lane & 15);
    int mo = (lane >> 4) * 8;
#pragma unroll
    for (int r = 0; r < 8; ++r) {
      long long node = n0 + rb + mo + r;
      if (node < (long long)N)
        out[node * 384 + n] = c[r];
    }
  }
}

// ---------------------------------------------------------------------------
// Host-side launcher.
// ---------------------------------------------------------------------------
extern "C" void kernel_launch(void* const* d_in, const int* in_sizes, int n_in,
                              void* d_out, int out_size, void* d_ws,
                              size_t ws_size, hipStream_t stream) {
  const float* x          = (const float*)d_in[0];
  const int*   eidx       = (const int*)d_in[1];
  const float* eattr      = (const float*)d_in[2];
  const float* u          = (const float*)d_in[3];
  const int*   node_batch = (const int*)d_in[4];
  // d_in[5] edge_batch: unused by reference computation
  const float* wts  = (const float*)d_in[6];
  const float* mw1  = (const float*)d_in[7];
  const float* mb1  = (const float*)d_in[8];
  const float* mg1  = (const float*)d_in[9];
  const float* mbe1 = (const float*)d_in[10];
  const float* mw2  = (const float*)d_in[11];
  const float* mb2  = (const float*)d_in[12];
  const float* nw1  = (const float*)d_in[13];
  const float* nb1  = (const float*)d_in[14];
  const float* ng1  = (const float*)d_in[15];
  const float* nbe1 = (const float*)d_in[16];
  const float* nw2  = (const float*)d_in[17];
  const float* nb2  = (const float*)d_in[18];
  float* out = (float*)d_out;

  const int N = in_sizes[0] / 128;   // nodes
  const int E = in_sizes[2] / 128;   // edges
  const int B = in_sizes[3] / 64;    // graphs

  const long long nx = (long long)N * 128;
  const long long ne = (long long)E * 128;
  const long long nu = (long long)B * 64;

  // Workspace carve-up.
  char* ws = (char*)d_ws;
  size_t off = 0;
  _Float16* W1m = (_Float16*)(ws + off); off += (size_t)256 * 256 * 2;
  _Float16* W2m = (_Float16*)(ws + off); off += (size_t)256 * 256 * 2;
  _Float16* W1n = (_Float16*)(ws + off); off += (size_t)448 * 512 * 2;
  _Float16* W2n = (_Float16*)(ws + off); off += (size_t)512 * 384 * 2;
  _Float16* xh  = (_Float16*)(ws + off); off += (size_t)nx * 2;
  _Float16* eah = (_Float16*)(ws + off); off += (size_t)ne * 2;
  _Float16* uh  = (_Float16*)(ws + off); off += (size_t)nu * 2;
  off = (off + 255) & ~(size_t)255;
  float* counts = (float*)(ws + off); off += (size_t)N * 4;
  off = (off + 255) & ~(size_t)255;
  float* sums   = (float*)(ws + off); off += (size_t)N * 256 * 4;
  (void)ws_size; (void)n_in; (void)out_size;

  hipMemsetAsync(counts, 0, (size_t)N * 4, stream);
  hipMemsetAsync(sums, 0, (size_t)N * 256 * 4, stream);

  // f32 -> f16 activation pre-pass (halves gather/stream traffic, enables DMA).
  f32_to_f16_kernel<<<(int)((nx + 255) / 256), 256, 0, stream>>>(x, xh, nx);
  f32_to_f16_kernel<<<(int)((ne + 255) / 256), 256, 0, stream>>>(eattr, eah, ne);
  f32_to_f16_kernel<<<(int)((nu + 255) / 256), 256, 0, stream>>>(u, uh, nu);

  // Weight conversion + B-fragment swizzle (f32 -> f16), L2-resident afterwards.
  {
    int k1 = 256 * 256, k2 = 448 * 512, k3 = 512 * 384;
    convert_swizzle_kernel<<<(k1 + 255) / 256, 256, 0, stream>>>(mw1, W1m, 256, 256);
    convert_swizzle_kernel<<<(k1 + 255) / 256, 256, 0, stream>>>(mw2, W2m, 256, 256);
    convert_swizzle_kernel<<<(k2 + 255) / 256, 256, 0, stream>>>(nw1, W1n, 448, 512);
    convert_swizzle_kernel<<<(k3 + 255) / 256, 256, 0, stream>>>(nw2, W2n, 512, 384);
  }

  counts_kernel<<<(E + 255) / 256, 256, 0, stream>>>(eidx, counts, E);

  // Edge message MLP + scatter: 66,816 B dynamic LDS.
  {
    size_t shm = (size_t)(64 * 256 + 64 * 256) * 2 + 64 * 20;
    hipFuncSetAttribute(reinterpret_cast<const void*>(edge_msg_kernel),
                        hipFuncAttributeMaxDynamicSharedMemorySize, (int)shm);
    edge_msg_kernel<<<(E + 63) / 64, 128, shm, stream>>>(
        xh, eidx, eah, wts, W1m, mb1, mg1, mbe1, W2m, mb2, sums, E);
  }

  // Node MLP: 123,392 B dynamic LDS (within CDNA5's 320 KB/WGP budget).
  {
    size_t shm = (size_t)64 * 448 * 2 + (size_t)64 * 512 * 2 + 512;
    hipFuncSetAttribute(reinterpret_cast<const void*>(node_mlp_kernel),
                        hipFuncAttributeMaxDynamicSharedMemorySize, (int)shm);
    node_mlp_kernel<<<(N + 63) / 64, 128, shm, stream>>>(
        xh, sums, counts, uh, node_batch, W1n, nb1, ng1, nbe1, W2n, nb2, out, N);
  }
}
